// QRNN_Model_27101243638530
// MI455X (gfx1250) — matrix operand order repaired and checked by reading, not compiled, not run on hardware
//
#include <hip/hip_runtime.h>
#include <cstdint>
#include <cstddef>

typedef __attribute__((ext_vector_type(16))) __bf16 v16bf;
typedef __attribute__((ext_vector_type(8)))  float  v8f;

// ---------------------------------------------------------------- helpers

static __device__ __forceinline__ unsigned short f2bf(float f) {
    unsigned int u = __float_as_uint(f);
    unsigned int r = (u + 0x7FFFu + ((u >> 16) & 1u)) >> 16;  // RNE
    return (unsigned short)r;
}

static __device__ __forceinline__ v8f wmma_bf16(v16bf a, v16bf b, v8f c) {
    return __builtin_amdgcn_wmma_f32_16x16x32_bf16(
        /*neg_a=*/false, a, /*neg_b=*/false, b,
        /*c_mod=*/(short)0, c, /*reuse_a=*/false, /*reuse_b=*/false);
}

// A-matrix fragment (16-bit A 16x32), contiguous k order matching fragB_row:
// lanes 0-15 : elems 0-15 = K0..15
// lanes 16-31: elems 0-15 = K16..31
static __device__ __forceinline__ v16bf fragA_row(const unsigned short* rowPtr, int k, int lane) {
    const unsigned short* p = rowPtr + k + ((lane >> 4) << 4);
    v16bf f;
    ((uint4*)&f)[0] = *(const uint4*)(p);
    ((uint4*)&f)[1] = *(const uint4*)(p + 8);
    return f;
}

// B-matrix fragment (16-bit B 32x16): lane n in 0-15 holds column n K0..15,
// lane n+16 holds column n K16..31 (contiguous 32B from row-major W row n).
static __device__ __forceinline__ v16bf fragB_row(const unsigned short* rowPtr, int k, int lane) {
    const unsigned short* p = rowPtr + k + ((lane >> 4) << 4);
    v16bf f;
    ((uint4*)&f)[0] = *(const uint4*)(p);
    ((uint4*)&f)[1] = *(const uint4*)(p + 8);
    return f;
}

// Async global->LDS DMA (CDNA5 GLOBAL_LOAD_ASYNC_TO_LDS_B128, tracked by ASYNCcnt).
// ldsPtr's generic address low 32 bits are the LDS byte offset (shared aperture layout).
static __device__ __forceinline__ void async_b128(unsigned short* ldsPtr,
                                                  const unsigned short* gptr) {
    unsigned int loff = (unsigned int)(uintptr_t)ldsPtr;
    asm volatile("global_load_async_to_lds_b128 %0, %1, off"
                 :: "v"(loff), "v"((unsigned long long)(uintptr_t)gptr)
                 : "memory");
}

static __device__ __forceinline__ void wait_async6() {
    asm volatile("s_wait_asynccnt 0x6" ::: "memory");
}
static __device__ __forceinline__ void wait_async0() {
    asm volatile("s_wait_asynccnt 0x0" ::: "memory");
}

// ---------------------------------------------------------------- pack kernels

// A0 row m = b*T+t: cols [0,INP) = x[b,t,:], cols [INP,2*INP) = x[b,t-1,:] (0 at t==0)
__global__ void pack_x_kernel(const float* __restrict__ x, unsigned short* __restrict__ A0,
                              int M, int T, int INP) {
    size_t n = (size_t)M * (2 * INP);
    for (size_t i = (size_t)blockIdx.x * blockDim.x + threadIdx.x; i < n;
         i += (size_t)gridDim.x * blockDim.x) {
        size_t m = i / (2 * INP);
        int    c = (int)(i % (2 * INP));
        float  v;
        if (c < INP) {
            v = x[m * INP + c];
        } else {
            int t = (int)(m % T);
            v = (t == 0) ? 0.f : x[(m - 1) * INP + (c - INP)];
        }
        A0[i] = f2bf(v);
    }
}

// Bcat row n: cols [0,K) = W[n,:], cols [K,2K) = V[n,:]
__global__ void pack_wcat_kernel(const float* __restrict__ W, const float* __restrict__ V,
                                 unsigned short* __restrict__ out, int N, int K) {
    size_t n = (size_t)N * 2 * K;
    for (size_t i = (size_t)blockIdx.x * blockDim.x + threadIdx.x; i < n;
         i += (size_t)gridDim.x * blockDim.x) {
        size_t row = i / (2 * K);
        int    c   = (int)(i % (2 * K));
        float  v   = (c < K) ? W[row * K + c] : V[row * K + (c - K)];
        out[i] = f2bf(v);
    }
}

__global__ void f32_to_bf16_kernel(const float* __restrict__ in,
                                   unsigned short* __restrict__ out, size_t n) {
    for (size_t i = (size_t)blockIdx.x * blockDim.x + threadIdx.x; i < n;
         i += (size_t)gridDim.x * blockDim.x)
        out[i] = f2bf(in[i]);
}

// ---------------------------------------------------------------- main WMMA GEMM
// C(MxN,f32) = A(MxK,bf16) * B(NxK,bf16)^T + bias(N)
// Block tile 128x256x32, 8 wave32 waves in 2(M) x 4(N), wave tile 64x64 (4x4 WMMA tiles).
// Global->LDS via async DMA (ASYNCcnt), double-buffered LDS stages.

#define BM 128
#define BN 256
#define BK 32
#define LDS_STRIDE 40  // 32 elems + 8 pad (80B pitch, 16B aligned)

__global__ __launch_bounds__(256)
void gemm_bias_f32(const unsigned short* __restrict__ A,
                   const unsigned short* __restrict__ Bm,
                   const float* __restrict__ bias,
                   float* __restrict__ C,
                   int M, int N, int K) {
    __shared__ __align__(16) unsigned short sA[2][BM * LDS_STRIDE];  // 2 x 10240 B
    __shared__ __align__(16) unsigned short sB[2][BN * LDS_STRIDE];  // 2 x 20480 B

    const int tid  = threadIdx.x;
    const int lane = tid & 31;
    const int wave = tid >> 5;           // 0..7
    const int wm   = (wave & 1) * 64;    // wave M offset in tile
    const int wn   = (wave >> 1) * 64;   // wave N offset in tile

    const int bm = blockIdx.y * BM;
    const int bn = blockIdx.x * BN;

    // per-wave async fill: lane -> (row = lane>>2, 16B chunk = lane&3); 512B per op
    const int fr = lane >> 2;            // 0..7
    const int fc = (lane & 3) << 3;      // 0,8,16,24 (ushort elems)

    v8f acc[4][4];
#pragma unroll
    for (int mi = 0; mi < 4; ++mi)
#pragma unroll
        for (int ni = 0; ni < 4; ++ni)
            acc[mi][ni] = (v8f){0.f, 0.f, 0.f, 0.f, 0.f, 0.f, 0.f, 0.f};

    const int nIter = K / BK;

    // issue one stage: 2 A ops (16 rows) + 4 B ops (32 rows) per wave = 6 async ops
    auto issue = [&](int stage, int k0) {
#pragma unroll
        for (int j = 0; j < 2; ++j) {
            int row = wave * 16 + j * 8 + fr;
            async_b128(&sA[stage][(size_t)row * LDS_STRIDE + fc],
                       A + (size_t)(bm + row) * K + k0 + fc);
        }
#pragma unroll
        for (int j = 0; j < 4; ++j) {
            int row = wave * 32 + j * 8 + fr;
            async_b128(&sB[stage][(size_t)row * LDS_STRIDE + fc],
                       Bm + (size_t)(bn + row) * K + k0 + fc);
        }
    };

    issue(0, 0);

    for (int i = 0; i < nIter; ++i) {
        const int s = i & 1;
        if (i + 1 < nIter) {
            issue(s ^ 1, (i + 1) * BK);   // overlap next fill with this compute
            wait_async6();                // first 6 (this stage) have landed
        } else {
            wait_async0();
        }
        __syncthreads();                  // stage s visible to all waves

        v16bf af[4], bfv[4];
#pragma unroll
        for (int mi = 0; mi < 4; ++mi)
            af[mi] = fragA_row(&sA[s][(size_t)(wm + mi * 16 + (lane & 15)) * LDS_STRIDE], 0, lane);
#pragma unroll
        for (int ni = 0; ni < 4; ++ni)
            bfv[ni] = fragB_row(&sB[s][(size_t)(wn + ni * 16 + (lane & 15)) * LDS_STRIDE], 0, lane);

#pragma unroll
        for (int mi = 0; mi < 4; ++mi)
#pragma unroll
            for (int ni = 0; ni < 4; ++ni)
                acc[mi][ni] = wmma_bf16(af[mi], bfv[ni], acc[mi][ni]);

        __syncthreads();                  // all waves done with stage s before refill
    }

    // epilogue: C/D layout (ISA 7.12.2): elem r -> M = r + (lane>=16 ? 8:0), N = lane&15
    const int colLane = lane & 15;
    const int rOff    = (lane >> 4) << 3;
#pragma unroll
    for (int mi = 0; mi < 4; ++mi) {
#pragma unroll
        for (int ni = 0; ni < 4; ++ni) {
            int   col  = bn + wn + ni * 16 + colLane;
            float bs   = bias[col];
            int   row0 = bm + wm + mi * 16 + rOff;
#pragma unroll
            for (int r = 0; r < 8; ++r)
                C[(size_t)(row0 + r) * N + col] = acc[mi][ni][r] + bs;
        }
    }
}

// ---------------------------------------------------------------- fused act + fo_pool scans

static __device__ __forceinline__ float sigm(float x) { return 1.f / (1.f + __expf(-x)); }

// layer0: read raw pre (B,T,3C), apply gates, scan, write h into A1 rows
// [cols 0..C) = h[t], cols [C..2C) = h[t-1] (zero at t==0)]
__global__ void scan0_kernel(const float* __restrict__ pre, unsigned short* __restrict__ A1,
                             int Bsz, int T, int C) {
    int g = blockIdx.x * blockDim.x + threadIdx.x;
    if (g >= Bsz * C) return;
    int b = g / C, ch = g % C;

    const size_t rs = (size_t)3 * C;
    const float* p  = pre + (size_t)b * T * rs + ch;
    unsigned short* a = A1 + (size_t)b * T * (2 * C);

    a[(size_t)C + ch] = f2bf(0.f);  // prev half at t==0
    float c = 0.f;
    for (int t = 0; t < T; ++t) {
        const float* q = p + (size_t)t * rs;
        float f = sigm(q[0]);
        float z = (1.f - f) * tanhf(q[C]);
        float o = sigm(q[2 * C]);
        c = c * f + z;
        unsigned short hb = f2bf(c * o);
        a[(size_t)t * (2 * C) + ch] = hb;
        if (t + 1 < T) a[(size_t)(t + 1) * (2 * C) + C + ch] = hb;
    }
}

// layer1: same scan but only the final timestep's h is needed (bf16 for the MLP)
__global__ void scan1_kernel(const float* __restrict__ pre, unsigned short* __restrict__ qb,
                             int Bsz, int T, int C) {
    int g = blockIdx.x * blockDim.x + threadIdx.x;
    if (g >= Bsz * C) return;
    int b = g / C, ch = g % C;

    const size_t rs = (size_t)3 * C;
    const float* p  = pre + (size_t)b * T * rs + ch;

    float c = 0.f, h = 0.f;
    for (int t = 0; t < T; ++t) {
        const float* q = p + (size_t)t * rs;
        float f = sigm(q[0]);
        float z = (1.f - f) * tanhf(q[C]);
        float o = sigm(q[2 * C]);
        c = c * f + z;
        h = c * o;
    }
    qb[(size_t)b * C + ch] = f2bf(h);
}

// ---------------------------------------------------------------- MLP head GEMM
// out(32xN) = A(32xK,bf16) * W(NxK,bf16)^T + bias; one wave per 16 output cols

__global__ __launch_bounds__(32)
void mlp_gemm_kernel(const unsigned short* __restrict__ A,
                     const unsigned short* __restrict__ W,
                     const float* __restrict__ bias,
                     unsigned short* __restrict__ outB, float* __restrict__ outF,
                     int N, int K, int doRelu) {
    int lane = threadIdx.x;
    int n0   = blockIdx.x * 16;

    v8f acc0 = (v8f){0.f, 0.f, 0.f, 0.f, 0.f, 0.f, 0.f, 0.f};
    v8f acc1 = acc0;

    int wrow = n0 + (lane & 15);
    if (wrow > N - 1) wrow = N - 1;  // clamp (dup rows are masked at store)
    const unsigned short* wPtr  = W + (size_t)wrow * K;
    const unsigned short* a0Ptr = A + (size_t)(lane & 15) * K;
    const unsigned short* a1Ptr = A + (size_t)(16 + (lane & 15)) * K;

    for (int k = 0; k < K; k += 32) {
        v16bf a0 = fragA_row(a0Ptr, k, lane);
        v16bf a1 = fragA_row(a1Ptr, k, lane);
        v16bf bb = fragB_row(wPtr, k, lane);
        acc0 = wmma_bf16(a0, bb, acc0);
        acc1 = wmma_bf16(a1, bb, acc1);
    }

    int col = n0 + (lane & 15);
    if (col < N) {
        float bs   = bias[col];
        int   rOff = (lane >> 4) << 3;
#pragma unroll
        for (int r = 0; r < 8; ++r) {
            float v0 = acc0[r] + bs;
            float v1 = acc1[r] + bs;
            if (doRelu) { v0 = v0 > 0.f ? v0 : 0.f; v1 = v1 > 0.f ? v1 : 0.f; }
            int row0 = rOff + r, row1 = 16 + rOff + r;
            if (outF) {
                outF[(size_t)row0 * N + col] = v0;
                outF[(size_t)row1 * N + col] = v1;
            } else {
                outB[(size_t)row0 * N + col] = f2bf(v0);
                outB[(size_t)row1 * N + col] = f2bf(v1);
            }
        }
    }
}

// ---------------------------------------------------------------- host launch

extern "C" void kernel_launch(void* const* d_in, const int* in_sizes, int n_in,
                              void* d_out, int out_size, void* d_ws, size_t ws_size,
                              hipStream_t stream) {
    const float* x   = (const float*)d_in[0];
    const float* W0  = (const float*)d_in[1];
    const float* V0  = (const float*)d_in[2];
    const float* b0  = (const float*)d_in[3];
    const float* W1  = (const float*)d_in[4];
    const float* V1  = (const float*)d_in[5];
    const float* b1  = (const float*)d_in[6];
    const float* l0w = (const float*)d_in[7];
    const float* l0b = (const float*)d_in[8];
    const float* l1w = (const float*)d_in[9];
    const float* l1b = (const float*)d_in[10];
    const float* l2w = (const float*)d_in[11];
    const float* l2b = (const float*)d_in[12];

    const int Bsz = 32, T = 1024, INP = 512, MODEL = 1024, MLPD = 1024, NCLS = 1000;
    const int M  = Bsz * T;     // 32768
    const int N3 = 3 * MODEL;   // 3072

    char*  ws  = (char*)d_ws;
    size_t off = 0;
    auto take = [&](size_t bytes) -> char* {
        char* p = ws + off;
        off = (off + bytes + 255) & ~(size_t)255;
        return p;
    };

    unsigned short* A0  = (unsigned short*)take((size_t)M * 2 * INP * 2);     // [X|Xprev] bf16
    unsigned short* B0c = (unsigned short*)take((size_t)N3 * 2 * INP * 2);    // [W0|V0]   bf16
    unsigned short* A1  = (unsigned short*)take((size_t)M * 2 * MODEL * 2);   // [H|Hprev] bf16
    unsigned short* B1c = (unsigned short*)take((size_t)N3 * 2 * MODEL * 2);  // [W1|V1]   bf16
    float*          pre = (float*)take((size_t)M * N3 * 4);                   // raw gates f32
    unsigned short* qb  = (unsigned short*)take((size_t)Bsz * MODEL * 2);
    unsigned short* w0b = (unsigned short*)take((size_t)MLPD * MODEL * 2);
    unsigned short* w1b = (unsigned short*)take((size_t)MLPD * MLPD * 2);
    unsigned short* w2b = (unsigned short*)take((size_t)NCLS * MLPD * 2);
    unsigned short* q1b = (unsigned short*)take((size_t)Bsz * MLPD * 2);
    unsigned short* q2b = (unsigned short*)take((size_t)Bsz * MLPD * 2);

    // ---- pack / convert
    pack_x_kernel<<<4096, 256, 0, stream>>>(x, A0, M, T, INP);
    pack_wcat_kernel<<<2048, 256, 0, stream>>>(W0, V0, B0c, N3, INP);
    pack_wcat_kernel<<<4096, 256, 0, stream>>>(W1, V1, B1c, N3, MODEL);
    f32_to_bf16_kernel<<<2048, 256, 0, stream>>>(l0w, w0b, (size_t)MLPD * MODEL);
    f32_to_bf16_kernel<<<2048, 256, 0, stream>>>(l1w, w1b, (size_t)MLPD * MLPD);
    f32_to_bf16_kernel<<<2048, 256, 0, stream>>>(l2w, w2b, (size_t)NCLS * MLPD);

    // ---- QRNN layer 0: fused dual GEMM (K = 2*INP) + fused act/scan/pack
    {
        dim3 grid(N3 / BN, M / BM);
        gemm_bias_f32<<<grid, 256, 0, stream>>>(A0, B0c, b0, pre, M, N3, 2 * INP);
    }
    scan0_kernel<<<(Bsz * MODEL + 255) / 256, 256, 0, stream>>>(pre, A1, Bsz, T, MODEL);

    // ---- QRNN layer 1 (K = 2*MODEL); only the last timestep survives the scan
    {
        dim3 grid(N3 / BN, M / BM);
        gemm_bias_f32<<<grid, 256, 0, stream>>>(A1, B1c, b1, pre, M, N3, 2 * MODEL);
    }
    scan1_kernel<<<(Bsz * MODEL + 255) / 256, 256, 0, stream>>>(pre, qb, Bsz, T, MODEL);

    // ---- MLP head
    mlp_gemm_kernel<<<MLPD / 16, 32, 0, stream>>>(qb, w0b, l0b, q1b, nullptr, MLPD, MODEL, 1);
    mlp_gemm_kernel<<<MLPD / 16, 32, 0, stream>>>(q1b, w1b, l1b, q2b, nullptr, MLPD, MLPD, 1);
    mlp_gemm_kernel<<<(NCLS + 15) / 16, 32, 0, stream>>>(q2b, w2b, l2b, nullptr, (float*)d_out,
                                                         NCLS, MLPD, 0);

    (void)in_sizes; (void)n_in; (void)out_size; (void)ws_size;
}
